// CausalSelfAttention_38311108280325
// MI455X (gfx1250) — matrix operand order repaired and checked by reading, not compiled, not run on hardware
//
#include <hip/hip_runtime.h>

// ---------------------------------------------------------------------------
// Causal self-attention forward for MI455X (gfx1250).
// bf16 WMMA compute, TDM (tensor_load_to_lds) double-buffered GEMM staging.
// B=2, T=2048, C=1024, H=16, d=64.
// ---------------------------------------------------------------------------

typedef __bf16 bf16_t;
typedef __attribute__((ext_vector_type(16))) __bf16 v16bf;
typedef __attribute__((ext_vector_type(8)))  __bf16 v8bf;
typedef __attribute__((ext_vector_type(8)))  float  v8f;
typedef __attribute__((ext_vector_type(4)))  unsigned int u32x4;
typedef __attribute__((ext_vector_type(8)))  int i32x8;
typedef __attribute__((ext_vector_type(4)))  int i32x4;

static __device__ __forceinline__ bf16_t f2bf(float f) {
  unsigned u = __builtin_bit_cast(unsigned, f);
  u += 0x7FFFu + ((u >> 16) & 1u);            // round-to-nearest-even
  unsigned short s = (unsigned short)(u >> 16);
  return __builtin_bit_cast(bf16_t, s);
}

// A-fragment (16x32 bf16, MxK): lane holds row m=lane%16; elements 0..15 are
// k = half*16+0..15  (half = lane/16).
static __device__ __forceinline__ v16bf load_afrag(const bf16_t* rowk0, int half) {
  union { v16bf v; v8bf h[2]; } u;
  u.h[0] = *(const v8bf*)(rowk0 + half * 16);
  u.h[1] = *(const v8bf*)(rowk0 + half * 16 + 8);
  return u.v;
}

// ---------------------------------------------------------------------------
// TDM: 2D tile load Global -> LDS via Tensor Data Mover (D# groups per
// CDNA5 ISA).  data_size=2B (bf16).  Optional LDS
// padding: pad_interval code 3 = every 64B, pad_amount code 3 = 16B pad,
// giving an 80B LDS row stride for bank-conflict-free ds_load_b128 reads.
// This toolchain exposes the 6-arg builtin:
//   (u32x4 g0, i32x8 g1, i32x4 g2, i32x4 g3, i32x8 g4, i32 cpol)
// ---------------------------------------------------------------------------
static __device__ __forceinline__ void tdm_load_2d(unsigned lds_off, const void* gptr,
                                                   unsigned tile_d0, unsigned tile_d1,
                                                   unsigned long long stride0, int pad_en) {
  unsigned long long ga = (unsigned long long)gptr;
  u32x4 g0;
  g0[0] = 1u;                                           // count=1, user mode
  g0[1] = lds_off;                                      // lds_addr (bytes)
  g0[2] = (unsigned)(ga & 0xffffffffu);                 // global_addr[31:0]
  g0[3] = (unsigned)((ga >> 32) & 0x1ffffffu)           // global_addr[56:32]
        | (2u << 30);                                   // type=2 ("image")
  i32x8 g1;
  unsigned w0 = (1u << 16);                             // data_size=1 -> 2 bytes
  if (pad_en) w0 |= (1u << 20) | (3u << 22) | (3u << 25);  // pad 64B->+16B
  g1[0] = (int)w0;
  g1[1] = (int)((tile_d0 & 0xffffu) << 16);             // tensor_dim0 = tile_d0
  g1[2] = (int)((tile_d0 >> 16) | ((tile_d1 & 0xffffu) << 16));  // tensor_dim1
  g1[3] = (int)((tile_d1 >> 16) | (tile_d0 << 16));     // tile_dim0
  g1[4] = (int)(tile_d1 & 0xffffu);                     // tile_dim1 (tile_dim2=0)
  g1[5] = (int)(unsigned)(stride0 & 0xffffffffu);       // tensor_dim0_stride lo
  g1[6] = (int)(unsigned)((stride0 >> 32) & 0xffffu);   // stride hi (dim1_stride=0)
  g1[7] = 0;
  i32x4 z4 = {0, 0, 0, 0};
  i32x8 z8 = {0, 0, 0, 0, 0, 0, 0, 0};
  __builtin_amdgcn_tensor_load_to_lds(g0, g1, z4, z4, z8, 0);
}

// ---------------------------------------------------------------------------
// Elementwise f32 -> bf16
// ---------------------------------------------------------------------------
__global__ void conv_f32_to_bf16(const float* __restrict__ in,
                                 bf16_t* __restrict__ out, int n) {
  int i = blockIdx.x * 256 + threadIdx.x;
  if (i < n) out[i] = f2bf(in[i]);
}

// W [K][N] f32 -> Wt [N][K] bf16  (so WMMA B-fragments are contiguous loads)
__global__ void transpose_f32_to_bf16(const float* __restrict__ W,
                                      bf16_t* __restrict__ Wt, int K, int N) {
  int n = blockIdx.x * 32 + (threadIdx.x & 31);
  int k = blockIdx.y * 8 + (threadIdx.x >> 5);
  if (n < N && k < K) Wt[(size_t)n * K + k] = f2bf(W[(size_t)k * N + n]);
}

// ---------------------------------------------------------------------------
// GEMM: Out = A[M,K](bf16) x Bt[N,K](bf16, pre-transposed) + bias.
// Block tile 128x128, 8 waves (2 along M x 4 along N), wave tile 64x32.
// K staged 32-deep into LDS by TDM, double buffered; 8 WMMAs/wave/stage.
// mode 0: store bf16 at [B*H, T, 64]   (Q, K head-split layout)
// mode 1: store bf16 at [B*H, 64, T]   (V transposed layout)
// mode 2: store f32  at [M, N]         (final projection -> d_out)
// ---------------------------------------------------------------------------
__global__ __launch_bounds__(256)
void gemm_bf16(const bf16_t* __restrict__ A, const bf16_t* __restrict__ Bt,
               const float* __restrict__ bias, void* __restrict__ Out,
               int M, int N, int K, int mode) {
  // 80-byte row stride (40 bf16) from TDM padding: bank-conflict free.
  __shared__ __align__(16) bf16_t As[2][128 * 40];
  __shared__ __align__(16) bf16_t Bs[2][128 * 40];

  const int lane = threadIdx.x & 31;
  const int wave = threadIdx.x >> 5;
  const int half = lane >> 4;
  const int lm   = lane & 15;
  const int wm   = wave & 1;
  const int wn   = wave >> 1;
  const int m0 = blockIdx.y * 128;
  const int n0 = blockIdx.x * 128;
  const int mw = wm * 64;
  const int nw = wn * 32;

  v8f acc[4][2] = {};

  const int nk = K >> 5;
  if (wave == 0) {                       // TDM ignores EXEC: one DMA per issue
    tdm_load_2d((unsigned)(uintptr_t)&As[0][0], A  + (size_t)m0 * K, 32, 128,
                (unsigned long long)K, 1);
    tdm_load_2d((unsigned)(uintptr_t)&Bs[0][0], Bt + (size_t)n0 * K, 32, 128,
                (unsigned long long)K, 1);
  }

  for (int kc = 0; kc < nk; ++kc) {
    const int cur = kc & 1;
    if (wave == 0) {
      if (kc + 1 < nk) {                 // prefetch next stage into other buffer
        const int k0n = (kc + 1) << 5;
        tdm_load_2d((unsigned)(uintptr_t)&As[cur ^ 1][0],
                    A + (size_t)m0 * K + k0n, 32, 128, (unsigned long long)K, 1);
        tdm_load_2d((unsigned)(uintptr_t)&Bs[cur ^ 1][0],
                    Bt + (size_t)n0 * K + k0n, 32, 128, (unsigned long long)K, 1);
        __builtin_amdgcn_s_wait_tensorcnt(2);   // stage `cur` complete
      } else {
        __builtin_amdgcn_s_wait_tensorcnt(0);
      }
    }
    __syncthreads();

    v16bf af[4], bfr[2];
#pragma unroll
    for (int i = 0; i < 4; ++i)
      af[i] = load_afrag(&As[cur][(mw + i * 16 + lm) * 40], half);
#pragma unroll
    for (int j = 0; j < 2; ++j) {
      const bf16_t* p = &Bs[cur][(nw + j * 16 + lm) * 40 + half * 16];
      union { v16bf v; v8bf h[2]; } u;
      u.h[0] = *(const v8bf*)(p);
      u.h[1] = *(const v8bf*)(p + 8);
      bfr[j] = u.v;
    }
#pragma unroll
    for (int i = 0; i < 4; ++i)
#pragma unroll
      for (int j = 0; j < 2; ++j)
        acc[i][j] = __builtin_amdgcn_wmma_f32_16x16x32_bf16(
            false, af[i], false, bfr[j], (short)0, acc[i][j], false, false);

    __syncthreads();                     // reads done before next TDM overwrite
  }

  // Epilogue: bias + store per mode
#pragma unroll
  for (int j = 0; j < 2; ++j) {
    const int c  = n0 + nw + j * 16 + lm;
    const float bv = bias[c];
#pragma unroll
    for (int i = 0; i < 4; ++i) {
#pragma unroll
      for (int r = 0; r < 8; ++r) {
        const int   m = m0 + mw + i * 16 + r + 8 * half;
        const float v = acc[i][j][r] + bv;
        if (mode == 2) {
          ((float*)Out)[(size_t)m * N + c] = v;
        } else {
          const int b_ = m >> 11;          // m / 2048  (T = 2048)
          const int t  = m & 2047;
          const int h  = c >> 6;
          const int dd = c & 63;
          const int bh = b_ * 16 + h;
          const size_t idx = (mode == 0)
              ? ((size_t)(bh * 2048 + t) * 64 + dd)
              : ((size_t)(bh * 64 + dd) * 2048 + t);
          ((bf16_t*)Out)[idx] = f2bf(v);
        }
      }
    }
  }
}

// ---------------------------------------------------------------------------
// Flash attention (causal). Grid: (T/64, B*H). Block: 128 thr = 4 waves.
// Each wave owns 16 query rows; loops over 32-key chunks with online softmax.
// Q,K: [B*H, T, 64] bf16.  Vt: [B*H, 64, T] bf16.  Y: [B*T, 1024] bf16.
// ---------------------------------------------------------------------------
__global__ __launch_bounds__(128)
void attn_fwd(const bf16_t* __restrict__ Q, const bf16_t* __restrict__ Kmat,
              const bf16_t* __restrict__ Vt, bf16_t* __restrict__ Y) {
  const int Tn = 2048, D = 64;
  __shared__ __align__(16) bf16_t plds[4][16 * 40];  // per-wave P staging, padded

  const int lane  = threadIdx.x & 31;
  const int wave  = threadIdx.x >> 5;
  const int half  = lane >> 4;
  const int lm    = lane & 15;
  const int bh    = blockIdx.y;
  const int qbase = blockIdx.x * 64 + wave * 16;

  const bf16_t* Qp = Q    + (size_t)bh * Tn * D;
  const bf16_t* Kp = Kmat + (size_t)bh * Tn * D;
  const bf16_t* Vp = Vt   + (size_t)bh * D * Tn;

  v16bf qa[2];
#pragma unroll
  for (int dc = 0; dc < 2; ++dc)
    qa[dc] = load_afrag(Qp + (size_t)(qbase + lm) * D + dc * 32, half);

  v8f o[4] = {};
  float mr[8], lr[8];
#pragma unroll
  for (int r = 0; r < 8; ++r) { mr[r] = -1e30f; lr[r] = 0.0f; }

  const int nch = (qbase + 16 + 31) >> 5;  // causal chunk count
  for (int kc = 0; kc < nch; ++kc) {
    const int kb = kc << 5;

    v8f s[2] = {};
#pragma unroll
    for (int j = 0; j < 2; ++j) {
      const int key = kb + j * 16 + lm;
#pragma unroll
      for (int dc = 0; dc < 2; ++dc) {
        v16bf kf = *(const v16bf*)(Kp + (size_t)key * D + dc * 32 + half * 16);
        s[j] = __builtin_amdgcn_wmma_f32_16x16x32_bf16(
            false, qa[dc], false, kf, (short)0, s[j], false, false);
      }
    }

    float p[2][8];
#pragma unroll
    for (int r = 0; r < 8; ++r) {
      const int q = qbase + r + 8 * half;
      float s0 = s[0][r] * 0.125f;                 // 1/sqrt(64)
      float s1 = s[1][r] * 0.125f;
      if (kb + lm      > q) s0 = -1e30f;
      if (kb + 16 + lm > q) s1 = -1e30f;
      float cmax = fmaxf(s0, s1);
#pragma unroll
      for (int d_ = 1; d_ < 16; d_ <<= 1)
        cmax = fmaxf(cmax, __shfl_xor(cmax, d_, 32));
      const float mn    = fmaxf(mr[r], cmax);
      const float alpha = __expf(mr[r] - mn);
      mr[r] = mn;
      const float p0 = __expf(s0 - mn);
      const float p1 = __expf(s1 - mn);
      float rs = p0 + p1;
#pragma unroll
      for (int d_ = 1; d_ < 16; d_ <<= 1) rs += __shfl_xor(rs, d_, 32);
      lr[r] = lr[r] * alpha + rs;
#pragma unroll
      for (int j4 = 0; j4 < 4; ++j4) o[j4][r] *= alpha;
      p[0][r] = p0;
      p[1][r] = p1;
    }

    // Transpose P (C-layout) -> A-fragment layout via wave-private LDS
    bf16_t* pw = plds[wave];
#pragma unroll
    for (int j = 0; j < 2; ++j)
#pragma unroll
      for (int r = 0; r < 8; ++r)
        pw[(r + 8 * half) * 40 + j * 16 + lm] = f2bf(p[j][r]);

    union { v16bf v; v8bf h[2]; } u;
    u.h[0] = *(const v8bf*)(pw + lm * 40 + half * 16);
    u.h[1] = *(const v8bf*)(pw + lm * 40 + half * 16 + 8);
    const v16bf pa = u.v;

#pragma unroll
    for (int j4 = 0; j4 < 4; ++j4) {
      v16bf vf = *(const v16bf*)(Vp + (size_t)(j4 * 16 + lm) * Tn + kb + half * 16);
      o[j4] = __builtin_amdgcn_wmma_f32_16x16x32_bf16(
          false, pa, false, vf, (short)0, o[j4], false, false);
    }
  }

  const int b_ = bh >> 4;
  const int h  = bh & 15;
#pragma unroll
  for (int j4 = 0; j4 < 4; ++j4)
#pragma unroll
    for (int r = 0; r < 8; ++r) {
      const int q = qbase + r + 8 * half;
      const float v = o[j4][r] / lr[r];
      Y[(size_t)(b_ * Tn + q) * 1024 + h * 64 + j4 * 16 + lm] = f2bf(v);
    }
}

// ---------------------------------------------------------------------------
// Host orchestration
// ---------------------------------------------------------------------------
extern "C" void kernel_launch(void* const* d_in, const int* in_sizes, int n_in,
                              void* d_out, int out_size, void* d_ws, size_t ws_size,
                              hipStream_t stream) {
  (void)in_sizes; (void)n_in; (void)out_size; (void)ws_size;
  const float* x  = (const float*)d_in[0];
  const float* Wq = (const float*)d_in[1];
  const float* bq = (const float*)d_in[2];
  const float* Wk = (const float*)d_in[3];
  const float* bk = (const float*)d_in[4];
  const float* Wv = (const float*)d_in[5];
  const float* bv = (const float*)d_in[6];
  const float* Wo = (const float*)d_in[7];
  const float* bo = (const float*)d_in[8];
  float* out = (float*)d_out;

  const int M = 4096, C = 1024;            // M = B*T
  char* ws = (char*)d_ws;
  const size_t MB = 1024 * 1024;
  bf16_t* xbf  = (bf16_t*)(ws + 0 * MB);   //  8 MB
  bf16_t* Wqt  = (bf16_t*)(ws + 8 * MB);   //  2 MB
  bf16_t* Wkt  = (bf16_t*)(ws + 10 * MB);  //  2 MB
  bf16_t* Wvt  = (bf16_t*)(ws + 12 * MB);  //  2 MB
  bf16_t* Wot  = (bf16_t*)(ws + 14 * MB);  //  2 MB
  bf16_t* Qbf  = (bf16_t*)(ws + 16 * MB);  //  8 MB  [B*H, T, 64]
  bf16_t* Kbf  = (bf16_t*)(ws + 24 * MB);  //  8 MB  [B*H, T, 64]
  bf16_t* Vtbf = (bf16_t*)(ws + 32 * MB);  //  8 MB  [B*H, 64, T]
  bf16_t* ybf  = (bf16_t*)(ws + 40 * MB);  //  8 MB  [B*T, C]

  const int nx = M * C;
  conv_f32_to_bf16<<<nx / 256, 256, 0, stream>>>(x, xbf, nx);

  dim3 tg(C / 32, C / 8);
  transpose_f32_to_bf16<<<tg, 256, 0, stream>>>(Wq, Wqt, C, C);
  transpose_f32_to_bf16<<<tg, 256, 0, stream>>>(Wk, Wkt, C, C);
  transpose_f32_to_bf16<<<tg, 256, 0, stream>>>(Wv, Wvt, C, C);
  transpose_f32_to_bf16<<<tg, 256, 0, stream>>>(Wo, Wot, C, C);

  dim3 gg(C / 128, M / 128);
  gemm_bf16<<<gg, 256, 0, stream>>>(xbf, Wqt, bq, Qbf,  M, C, C, 0);
  gemm_bf16<<<gg, 256, 0, stream>>>(xbf, Wkt, bk, Kbf,  M, C, C, 0);
  gemm_bf16<<<gg, 256, 0, stream>>>(xbf, Wvt, bv, Vtbf, M, C, C, 1);

  attn_fwd<<<dim3(32, 32), 128, 0, stream>>>(Qbf, Kbf, Vtbf, ybf);

  gemm_bf16<<<gg, 256, 0, stream>>>(ybf, Wot, bo, out, M, C, C, 2);
}
